// Attention_44590350467591
// MI455X (gfx1250) — matrix operand order repaired and checked by reading, not compiled, not run on hardware
//
#include <hip/hip_runtime.h>
#include <hip/hip_bf16.h>

// ---------------------------------------------------------------------------
// Problem constants (fixed by the reference)
// ---------------------------------------------------------------------------
#define BB      2
#define NQ      2048          // query sequence length
#define MK      2048          // context sequence length
#define DMODEL  1024
#define HEADS   16
#define DH      64
#define SCALE_Q 0.125f        // 1/sqrt(64)
#define RMS_SC  32.0f         // sqrt(1024)

typedef _Float16 v16h __attribute__((ext_vector_type(16)));
typedef float    v8f  __attribute__((ext_vector_type(8)));
typedef int      v4i  __attribute__((vector_size(16)));
typedef __attribute__((address_space(1))) v4i* gv4p;   // global v4i*
typedef __attribute__((address_space(3))) v4i* lv4p;   // LDS v4i*

union AF { v16h v; uint4 q[2]; };

__device__ __forceinline__ v8f wmma_f16(v16h a, v16h b, v8f c) {
    // D = A(16x32 f16) * B(32x16 f16) + C(16x16 f32)
    return __builtin_amdgcn_wmma_f32_16x16x32_f16(
        false, a, false, b, (short)0, c, false, false);
}

#if __has_builtin(__builtin_amdgcn_global_load_async_to_lds_b128)
#define HAVE_ASYNC_LDS 1
__device__ __forceinline__ void async_cp16(const _Float16* g, _Float16* l) {
    // AS1 v4i* source (casts away const via integer), AS3 v4i* dest
    // (low 32 bits of a generic LDS address are the LDS byte offset).
    __builtin_amdgcn_global_load_async_to_lds_b128(
        (gv4p)(unsigned long long)g,
        (lv4p)(unsigned)(unsigned long long)l,
        0, 0);
}
#else
#define HAVE_ASYNC_LDS 0
__device__ __forceinline__ void async_cp16(const _Float16* g, _Float16* l) {
    *(uint4*)l = *(const uint4*)g;          // synchronous fallback
}
#endif

// ---------------------------------------------------------------------------
// Kernel 1: weight fp32 -> f16 with transpose.  WT[n*K + k] = (f16)W[k*N + n]
// ---------------------------------------------------------------------------
__global__ void wcast_t(const float* __restrict__ W, _Float16* __restrict__ WT,
                        int Kdim, int Ncols) {
    int idx = blockIdx.x * 256 + threadIdx.x;
    int total = Kdim * Ncols;
    if (idx < total) {
        int n = idx / Kdim;
        int k = idx - n * Kdim;
        WT[idx] = (_Float16)W[(size_t)k * Ncols + n];
    }
}

// ---------------------------------------------------------------------------
// Kernel 2: RMS-norm one row of 1024 floats -> f16
// ---------------------------------------------------------------------------
__global__ __launch_bounds__(256)
void rmsnorm_cast(const float* __restrict__ x, const float* __restrict__ gamma,
                  _Float16* __restrict__ out) {
    int row = blockIdx.x;                       // 0 .. B*N-1
    const float* xr = x + (size_t)row * DMODEL;
    float v[4];
    float ss = 0.f;
#pragma unroll
    for (int i = 0; i < 4; ++i) {
        v[i] = xr[threadIdx.x + i * 256];
        ss += v[i] * v[i];
    }
#pragma unroll
    for (int m = 1; m < 32; m <<= 1) ss += __shfl_xor(ss, m, 32);
    __shared__ float red[8];
    if ((threadIdx.x & 31) == 0) red[threadIdx.x >> 5] = ss;
    __syncthreads();
    float tot = red[0];
#pragma unroll
    for (int i = 1; i < 8; ++i) tot += red[i];
    float s = RMS_SC / fmaxf(sqrtf(tot), 1e-12f);
#pragma unroll
    for (int i = 0; i < 4; ++i) {
        int c = threadIdx.x + i * 256;
        out[(size_t)row * DMODEL + c] = (_Float16)(v[i] * s * gamma[c]);
    }
}

// ---------------------------------------------------------------------------
// Kernel 3: WMMA GEMM  C[Mrows x Nout] = A[Mrows x K] * WT[Nout x K]
// 256 threads = 8 waves; tile 128x128; each wave 32x64 (2x4 of 16x16).
// mode 0: Q epilogue; mode 1: KV epilogue; mode 2: fp32 output
// ---------------------------------------------------------------------------
__global__ __launch_bounds__(256)
void gemm_wmma(const _Float16* __restrict__ A, const _Float16* __restrict__ WT,
               int Mrows, int K, int Nout, int mode,
               _Float16* __restrict__ outH, _Float16* __restrict__ outH2,
               float* __restrict__ outF, const float* __restrict__ scores) {
    const int lane = threadIdx.x & 31;
    const int wave = threadIdx.x >> 5;
    const int lh   = lane & 15;
    const int hi   = lane >> 4;            // half-wave select
    const int wm   = wave >> 1;            // 0..3
    const int wn   = wave & 1;             // 0..1
    const int row0 = blockIdx.x * 128 + wm * 32;
    const int col0 = blockIdx.y * 128 + wn * 64;

    v8f acc[2][4] = {};

    for (int kb = 0; kb < K; kb += 32) {
        AF af[2];
#pragma unroll
        for (int i = 0; i < 2; ++i) {
            const _Float16* ap =
                A + (size_t)(row0 + i * 16 + lh) * K + kb + hi * 16;
            af[i].q[0] = *(const uint4*)ap;         // K chunk [k0, k0+8)
            af[i].q[1] = *(const uint4*)(ap + 8);   // K chunk [k0+8, k0+16)
            __builtin_prefetch((const void*)(ap + 32), 0, 3);  // speculative
        }
        AF bf[4];
#pragma unroll
        for (int j = 0; j < 4; ++j) {
            const _Float16* bp =
                WT + (size_t)(col0 + j * 16 + lh) * K + kb + hi * 16;
            bf[j].v = *(const v16h*)bp;             // 16 contiguous K halves
            __builtin_prefetch((const void*)(bp + 32), 0, 3);
        }
#pragma unroll
        for (int i = 0; i < 2; ++i)
#pragma unroll
            for (int j = 0; j < 4; ++j)
                acc[i][j] = wmma_f16(af[i].v, bf[j].v, acc[i][j]);
    }

    // ---- epilogue: C layout lane L, VGPR e -> row e + 8*hi, col = L&15 ----
#pragma unroll
    for (int i = 0; i < 2; ++i) {
#pragma unroll
        for (int j = 0; j < 4; ++j) {
#pragma unroll
            for (int e = 0; e < 8; ++e) {
                int row = row0 + i * 16 + e + hi * 8;
                int col = col0 + j * 16 + lh;
                float val = acc[i][j][e];
                if (mode == 0) {                      // Q projection
                    int b = row >> 11, n = row & (NQ - 1);
                    int h = col >> 6,  d = col & (DH - 1);
                    outH[(((size_t)(b * HEADS + h) * NQ + n) * DH) + d] =
                        (_Float16)(val * SCALE_Q);
                } else if (mode == 1) {               // KV projection
                    int b = row >> 11, m = row & (MK - 1);
                    if (col < DMODEL) {               // K
                        int h = col >> 6, d = col & (DH - 1);
                        outH[(((size_t)(b * HEADS + h) * MK + m) * DH) + d] =
                            (_Float16)val;
                    } else {                          // V (scaled, transposed)
                        int c2 = col - DMODEL;
                        int h = c2 >> 6, d = c2 & (DH - 1);
                        float sc = scores[(size_t)b * MK + m];
                        outH2[(((size_t)(b * HEADS + h) * DH + d) * MK) + m] =
                            (_Float16)(val * sc);
                    }
                } else {                              // final fp32 output
                    outF[(size_t)row * Nout + col] = val;
                }
            }
        }
    }
}

// ---------------------------------------------------------------------------
// Kernel 4: flash attention with double-buffered async K/V staging in LDS.
// grid = (NQ/128, B*HEADS); 8 waves, 16 q-rows/wave.
// q16 [bh][n][64], k16 [bh][m][64], v16T [bh][64][m].
// ---------------------------------------------------------------------------
__global__ __launch_bounds__(256)
void flash_attn(const _Float16* __restrict__ q16,
                const _Float16* __restrict__ k16,
                const _Float16* __restrict__ v16T,
                const unsigned char* __restrict__ mask,
                _Float16* __restrict__ aout) {
    const int lane = threadIdx.x & 31;
    const int wave = threadIdx.x >> 5;
    const int lh   = lane & 15;
    const int hi   = lane >> 4;
    const int bh   = blockIdx.y;
    const int b    = bh >> 4;
    const int h    = bh & (HEADS - 1);
    const int q0   = blockIdx.x * 128 + wave * 16;

    const _Float16* qbase = q16 + ((size_t)bh * NQ + q0) * DH;
    const _Float16* kbase = k16 + (size_t)bh * MK * DH;
    const _Float16* vbase = v16T + (size_t)bh * DH * MK;
    const unsigned char* mbp = mask + (size_t)b * MK;

    // K tile: 32 keys x 64 d (4KB);  V tile: 64 d x 32 keys (4KB); both x2
    __shared__ __align__(64) _Float16 kbuf[2][32 * 64];
    __shared__ __align__(64) _Float16 vbuf[2][64 * 32];
    __shared__ __align__(64) _Float16 plds[8][16 * 32];   // per-wave P staging
    _Float16* myp = plds[wave];

    // per-thread cooperative-copy coordinates (16B granules)
    const int t   = threadIdx.x;
    const int kjl = t >> 3, kc = (t & 7) * 8;   // K: 32 rows x 8 granules
    const int vdl = t >> 2, vc = (t & 3) * 8;   // V: 64 rows x 4 granules

    auto stage = [&](int j0, int buf) {
        async_cp16(kbase + (size_t)(j0 + kjl) * DH + kc,
                   &kbuf[buf][kjl * 64 + kc]);
        async_cp16(vbase + (size_t)vdl * MK + j0 + vc,
                   &vbuf[buf][vdl * 32 + vc]);
    };

    // Q fragments for d-chunks {0,32}: resident whole kernel
    AF qf[2];
#pragma unroll
    for (int tq = 0; tq < 2; ++tq) {
        const _Float16* qp = qbase + (size_t)lh * DH + tq * 32 + hi * 16;
        qf[tq].q[0] = *(const uint4*)qp;
        qf[tq].q[1] = *(const uint4*)(qp + 8);
    }

    v8f o[4] = {};
    float mrun[8], lrun[8];
#pragma unroll
    for (int e = 0; e < 8; ++e) { mrun[e] = -3.0e38f; lrun[e] = 0.f; }

    const v8f z = {};

    stage(0, 0);                                  // prologue fill
    for (int j0 = 0, it = 0; j0 < MK; j0 += 32, ++it) {
        const int cur = it & 1;
        const bool more = (j0 + 32) < MK;
        if (more) stage(j0 + 32, cur ^ 1);        // prefetch next tile
#if HAVE_ASYNC_LDS
        if (more) asm volatile("s_wait_asynccnt 0x2" ::: "memory");
        else      asm volatile("s_wait_asynccnt 0x0" ::: "memory");
#endif
        __syncthreads();                          // tile visible to all waves

        const _Float16* kb_ = kbuf[cur];
        const _Float16* vb_ = vbuf[cur];

        // ---- S = q @ k^T for 32 keys (two 16-wide C tiles) ----
        const _Float16* kp0 = kb_ + (size_t)lh * DH + hi * 16;
        const _Float16* kp1 = kb_ + (size_t)(16 + lh) * DH + hi * 16;
        v8f s0 = wmma_f16(qf[0].v, *(const v16h*)kp0, z);
        s0     = wmma_f16(qf[1].v, *(const v16h*)(kp0 + 32), s0);
        v8f s1 = wmma_f16(qf[0].v, *(const v16h*)kp1, z);
        s1     = wmma_f16(qf[1].v, *(const v16h*)(kp1 + 32), s1);

        // ---- mask bias (per-lane column) ----
        float mb0 = mbp[j0 + lh]      ? 0.f : -3.0e38f;
        float mb1 = mbp[j0 + 16 + lh] ? 0.f : -3.0e38f;
#pragma unroll
        for (int e = 0; e < 8; ++e) { s0[e] += mb0; s1[e] += mb1; }

        // ---- online softmax (row reductions within each half-wave) ----
        float rmax[8];
#pragma unroll
        for (int e = 0; e < 8; ++e) rmax[e] = fmaxf(s0[e], s1[e]);
#pragma unroll
        for (int m = 1; m < 16; m <<= 1)
#pragma unroll
            for (int e = 0; e < 8; ++e)
                rmax[e] = fmaxf(rmax[e], __shfl_xor(rmax[e], m, 32));

        float sc[8], p0[8], p1[8], rsum[8];
#pragma unroll
        for (int e = 0; e < 8; ++e) {
            float nm = fmaxf(fmaxf(mrun[e], rmax[e]), -1.0e30f);
            sc[e]   = __expf(mrun[e] - nm);
            p0[e]   = __expf(s0[e] - nm);
            p1[e]   = __expf(s1[e] - nm);
            rsum[e] = p0[e] + p1[e];
            mrun[e] = nm;
        }
#pragma unroll
        for (int m = 1; m < 16; m <<= 1)
#pragma unroll
            for (int e = 0; e < 8; ++e) rsum[e] += __shfl_xor(rsum[e], m, 32);
#pragma unroll
        for (int e = 0; e < 8; ++e) lrun[e] = lrun[e] * sc[e] + rsum[e];
#pragma unroll
        for (int tt = 0; tt < 4; ++tt)
#pragma unroll
            for (int e = 0; e < 8; ++e) o[tt][e] *= sc[e];

        // ---- stage P (C layout) into LDS, reload as A fragment ----
#pragma unroll
        for (int e = 0; e < 8; ++e) {
            int row = e + hi * 8;
            myp[row * 32 + lh]      = (_Float16)p0[e];
            myp[row * 32 + 16 + lh] = (_Float16)p1[e];
        }
        asm volatile("s_wait_dscnt 0" ::: "memory");

        AF pf;
        {
            const _Float16* pp = myp + (size_t)lh * 32 + hi * 16;
            pf.q[0] = *(const uint4*)pp;
            pf.q[1] = *(const uint4*)(pp + 8);
        }
        asm volatile("s_wait_dscnt 0" ::: "memory");

        // ---- O += P @ V  (V rows contiguous in LDS [d][m] tile) ----
#pragma unroll
        for (int tt = 0; tt < 4; ++tt) {
            const _Float16* vp = vb_ + (size_t)(tt * 16 + lh) * 32 + hi * 16;
            o[tt] = wmma_f16(pf.v, *(const v16h*)vp, o[tt]);
        }

        __syncthreads();                 // everyone done reading this buffer
    }

    // ---- normalize and write [b, n, h*64 + d] as f16 for final GEMM ----
    float inv[8];
#pragma unroll
    for (int e = 0; e < 8; ++e) inv[e] = 1.f / fmaxf(lrun[e], 1e-30f);
#pragma unroll
    for (int tt = 0; tt < 4; ++tt)
#pragma unroll
        for (int e = 0; e < 8; ++e) {
            int row = q0 + e + hi * 8;
            int col = h * DH + tt * 16 + lh;
            aout[((size_t)(b * NQ + row)) * DMODEL + col] =
                (_Float16)(o[tt][e] * inv[e]);
        }
}

// ---------------------------------------------------------------------------
// Host launch
// ---------------------------------------------------------------------------
extern "C" void kernel_launch(void* const* d_in, const int* in_sizes, int n_in,
                              void* d_out, int out_size, void* d_ws,
                              size_t ws_size, hipStream_t stream) {
    (void)in_sizes; (void)n_in; (void)out_size; (void)ws_size;
    const float* x      = (const float*)d_in[0];
    const float* ctx    = (const float*)d_in[1];
    const float* scores = (const float*)d_in[2];
    const unsigned char* mask = (const unsigned char*)d_in[3];
    const float* gamma  = (const float*)d_in[4];
    const float* Wq     = (const float*)d_in[5];
    const float* Wkv    = (const float*)d_in[6];
    const float* Wout   = (const float*)d_in[7];

    const size_t ROWS = (size_t)BB * NQ;          // 4096
    _Float16* p = (_Float16*)d_ws;
    _Float16* xn16  = p; p += ROWS * DMODEL;
    _Float16* cn16  = p; p += ROWS * DMODEL;
    _Float16* WqT   = p; p += (size_t)DMODEL * DMODEL;
    _Float16* WkvT  = p; p += (size_t)2 * DMODEL * DMODEL;
    _Float16* WoutT = p; p += (size_t)DMODEL * DMODEL;
    _Float16* q16   = p; p += (size_t)BB * HEADS * NQ * DH;
    _Float16* k16   = p; p += (size_t)BB * HEADS * MK * DH;
    _Float16* v16T  = p; p += (size_t)BB * HEADS * DH * MK;
    _Float16* aout  = p;

    // 1. weights -> f16, transposed [N,K]
    wcast_t<<<4096, 256, 0, stream>>>(Wq,   WqT,   DMODEL, DMODEL);
    wcast_t<<<8192, 256, 0, stream>>>(Wkv,  WkvT,  DMODEL, 2 * DMODEL);
    wcast_t<<<4096, 256, 0, stream>>>(Wout, WoutT, DMODEL, DMODEL);

    // 2. RMS norm + cast
    rmsnorm_cast<<<(int)ROWS, 256, 0, stream>>>(x,   gamma, xn16);
    rmsnorm_cast<<<(int)ROWS, 256, 0, stream>>>(ctx, gamma, cn16);

    // 3. Q / KV projections (WMMA)
    gemm_wmma<<<dim3(32, 8),  256, 0, stream>>>(xn16, WqT,  (int)ROWS, DMODEL,
                                                DMODEL, 0, q16, nullptr,
                                                nullptr, nullptr);
    gemm_wmma<<<dim3(32, 16), 256, 0, stream>>>(cn16, WkvT, (int)ROWS, DMODEL,
                                                2 * DMODEL, 1, k16, v16T,
                                                nullptr, scores);

    // 4. flash attention (async double-buffered K/V staging)
    flash_attn<<<dim3(NQ / 128, BB * HEADS), 256, 0, stream>>>(q16, k16, v16T,
                                                               mask, aout);

    // 5. output projection -> fp32
    gemm_wmma<<<dim3(32, 8), 256, 0, stream>>>(aout, WoutT, (int)ROWS, DMODEL,
                                               DMODEL, 2, nullptr, nullptr,
                                               (float*)d_out, nullptr);
}
